// EinMask_81922206203994
// MI455X (gfx1250) — matrix operand order repaired and checked by reading, not compiled, not run on hardware
//
#include <hip/hip_runtime.h>

// ---------------- model constants ----------------
constexpr int kB    = 4;
constexpr int kTH   = 64, kTW = 64;
constexpr int kNT   = kTH * kTW;    // 4096 tokens
constexpr int kD    = 512;
constexpr int kFF   = 2048;
constexpr int kL    = 4;
constexpr int kCB   = 2;
constexpr int kH    = 8;
constexpr int kDH   = 64;           // head dim
constexpr int kNL   = 512;          // latents
constexpr int kS    = 2048;         // gathered context
constexpr int kK    = 2;
constexpr int kV    = 3;
constexpr int kPH   = 4, kPW = 4;

typedef unsigned short ushort_t;

// ---------------- WMMA / TDM types ----------------
typedef __bf16 v16bf __attribute__((ext_vector_type(16)));
typedef float  v8f   __attribute__((ext_vector_type(8)));
typedef unsigned int u32x4 __attribute__((ext_vector_type(4)));
typedef int          i32x8 __attribute__((ext_vector_type(8)));
typedef int          i32x4 __attribute__((ext_vector_type(4)));

#if defined(__has_builtin)
#if __has_builtin(__builtin_amdgcn_tensor_load_to_lds) && \
    __has_builtin(__builtin_amdgcn_s_wait_tensorcnt)
#define USE_TDM 1
#endif
#endif
#ifndef USE_TDM
#define USE_TDM 0
#endif

__device__ inline ushort_t f32_bf16(float f) {
  unsigned int u = __float_as_uint(f);
  unsigned int r = u + 0x7FFFu + ((u >> 16) & 1u);   // RNE
  return (ushort_t)(r >> 16);
}

__device__ inline v8f wmma_bf16(v16bf a, v16bf b, v8f c) {
  return __builtin_amdgcn_wmma_f32_16x16x32_bf16(false, a, false, b,
                                                 (short)0, c, false, false);
}

__device__ inline float gelu_f(float x) {
  const float c0 = 0.7978845608028654f;  // sqrt(2/pi)
  float t = tanhf(c0 * (x + 0.044715f * x * x * x));
  return 0.5f * x * (1.0f + t);
}

union FragQ { v16bf v; uint4 q[2]; unsigned int w[8]; ushort_t s[16]; };

// ---------------- fp32 -> bf16 conversion pass ----------------
__global__ void cvt_bf16_kernel(const float* __restrict__ x,
                                ushort_t* __restrict__ y, long n) {
  long idx = (long)blockIdx.x * blockDim.x + threadIdx.x;
  if (idx < n) y[idx] = f32_bf16(x[idx]);
}

// ---------------- LDS-staged batched WMMA GEMM (bf16 in) ----------------
// C[bz] = epi(A[bz] @ B[bz]); flags: 1=GELU, 2=+R(fp32 out), 4=bf16 out
// A: [M,K] bf16 row-major, B: [K,N] bf16 row-major. M%128==0, N%64==0, K%32==0.
// A panel (128x32, 8KB) staged by the Tensor Data Mover (wave 0 issues the
// descriptor, TENSORcnt-waits, workgroup barrier publishes); B panel (32x64,
// 4KB) staged by all threads with a writer-side transpose into WMMA fragment
// order.
__global__ __launch_bounds__(256) void gemm_wmma_kernel(
    const ushort_t* __restrict__ A, const ushort_t* __restrict__ Bw,
    const float* __restrict__ R, void* __restrict__ Cv,
    int M, int Nn, int K, long sA, long sB, long sR, long sC, int flags) {
  int bz = blockIdx.z;
  A  += (long)bz * sA;
  Bw += (long)bz * sB;
  const float* Rb = R ? (R + (long)bz * sR) : nullptr;

  int tid  = threadIdx.x;
  int lane = tid & 31;
  int wid  = tid >> 5;
  int wm = wid >> 1, wn = wid & 1;
  int mblk = blockIdx.y * 128;
  int nblk = blockIdx.x * 64;

  // A panel, row-major: row r occupies 64B; an A-fragment for a lane is two
  // b128 reads at row*64 + half*16 and row*64 + 32 + half*16.
  __shared__ __align__(16) ushort_t As[128 * 32];
  // B panel in fragment order: ((g*2+((k>>3)&1))*16 + n)*16 + (k&7) + 8*(k>>4), g = 16-col group
  __shared__ __align__(16) ushort_t Bs[4 * 2 * 16 * 16];

  // staging roles (B panel + fallback A panel)
  int arow = tid >> 1;               // 0..127
  int acp  = (tid & 1) * 2;          // chunk pair {0,1} or {2,3}
  int krow = tid & 31;               // 0..31
  int nc   = tid >> 5;               // 0..7 (8-col chunk)

  v8f acc00 = {}, acc01 = {}, acc10 = {}, acc11 = {};

  for (int k0 = 0; k0 < K; k0 += 32) {
    __syncthreads();  // previous iteration's fragment reads complete
    // ---- stage A panel ----
#if USE_TDM
    if (wid == 0) {
      unsigned lds_a =
          (unsigned)(unsigned long long)(__attribute__((address_space(3))) ushort_t*)As;
      unsigned long long ga =
          (unsigned long long)(const void*)(A + (long)mblk * K + k0);
      u32x4 g0 = {1u,                      // count=1 valid descriptor
                  lds_a,                   // lds_addr (bytes)
                  (unsigned)ga,            // global_addr[31:0]
                  (unsigned)((ga >> 32) & 0x01FFFFFFu) | (2u << 30)}; // type=2
      i32x8 g1;
      g1[0] = (int)(1u << 16);             // data_size = 2 bytes
      g1[1] = (int)(((unsigned)K & 0xFFFFu) << 16);                 // tensor_dim0 lo
      g1[2] = (int)(((unsigned)K >> 16) | (((unsigned)M & 0xFFFFu) << 16));
      g1[3] = (int)(((unsigned)M >> 16) | (32u << 16));             // tile_dim0=32
      g1[4] = 128;                         // tile_dim1 = 128 rows (tile_dim2=0)
      g1[5] = K;                           // tensor_dim0_stride
      g1[6] = 0;
      g1[7] = 0;
      i32x4 gz4 = {0, 0, 0, 0};
      i32x8 gz8 = {0, 0, 0, 0, 0, 0, 0, 0};
      __builtin_amdgcn_tensor_load_to_lds(g0, g1, gz4, gz4, gz8, 0);
    }
#else
    {
      const ushort_t* ga = A + (long)(mblk + arow) * K + k0;
      uint4* dst = (uint4*)((char*)As + arow * 64);
      dst[acp]     = *(const uint4*)(ga + acp * 8);
      dst[acp + 1] = *(const uint4*)(ga + acp * 8 + 8);
    }
#endif
    // ---- stage B (1x b128 load + 8x b16 LDS scatter per thread) ----
    {
      const ushort_t* gb = Bw + (long)(k0 + krow) * Nn + nblk + nc * 8;
      if (k0 + 32 < K) __builtin_prefetch(gb + 32 * Nn, 0, 3);
      FragQ d; d.q[0] = *(const uint4*)gb;
#pragma unroll
      for (int i = 0; i < 8; ++i) {
        int n = nc * 8 + i;
        Bs[(((n >> 4) * 2 + ((krow >> 3) & 1)) * 16 + (n & 15)) * 16 + (krow & 7) + ((krow >> 4) << 3)] = d.s[i];
      }
    }
#if USE_TDM
    if (wid == 0) __builtin_amdgcn_s_wait_tensorcnt(0);
#endif
    __syncthreads();
    // ---- fragments from LDS (2x ds_load_b128 each) ----
    FragQ a0, a1, b0, b1;
    {
      const char* asb = (const char*)As;
      int r0   = wm * 32 + (lane & 15);
      int hoff = (lane >> 4) * 16;
      a0.q[0] = *(const uint4*)(asb + r0 * 64 + hoff);
      a0.q[1] = *(const uint4*)(asb + r0 * 64 + 32 + hoff);
      a1.q[0] = *(const uint4*)(asb + (r0 + 16) * 64 + hoff);
      a1.q[1] = *(const uint4*)(asb + (r0 + 16) * 64 + 32 + hoff);
      int g0i = wn * 2;
      const uint4* q0 = (const uint4*)(Bs + ((g0i * 2 + (lane >> 4)) * 16 + (lane & 15)) * 16);
      b0.q[0] = q0[0]; b0.q[1] = q0[1];
      const uint4* q1 = (const uint4*)(Bs + (((g0i + 1) * 2 + (lane >> 4)) * 16 + (lane & 15)) * 16);
      b1.q[0] = q1[0]; b1.q[1] = q1[1];
    }
    acc00 = wmma_bf16(a0.v, b0.v, acc00);
    acc01 = wmma_bf16(a0.v, b1.v, acc01);
    acc10 = wmma_bf16(a1.v, b0.v, acc10);
    acc11 = wmma_bf16(a1.v, b1.v, acc11);
  }

  int mhalf = lane >> 4;
  int ncol  = lane & 15;
  float* Cf = (float*)Cv + (long)bz * sC;
  ushort_t* Ch = (ushort_t*)Cv + (long)bz * sC;
#pragma unroll
  for (int i = 0; i < 2; ++i) {
#pragma unroll
    for (int jn = 0; jn < 2; ++jn) {
      v8f a = (i == 0) ? (jn == 0 ? acc00 : acc01) : (jn == 0 ? acc10 : acc11);
#pragma unroll
      for (int r = 0; r < 8; ++r) {
        int row = mblk + wm * 32 + i * 16 + r + 8 * mhalf;
        int col = nblk + wn * 32 + jn * 16 + ncol;
        float val = a[r];
        if (flags & 1) val = gelu_f(val);
        if (flags & 2) val += Rb[(long)row * Nn + col];
        if (flags & 4) Ch[(long)row * Nn + col] = f32_bf16(val);
        else           Cf[(long)row * Nn + col] = val;
      }
    }
  }
}

// ---------------- bf16 fragment loaders for attention ----------------
__device__ inline v16bf load_a_frag_h(const ushort_t* __restrict__ base,
                                      int m0, int k0, long sm) {
  int lane = threadIdx.x & 31;
  int m = m0 + (lane & 15);
  int half = lane >> 4;
  FragQ u;
#pragma unroll
  for (int j = 0; j < 8; ++j) {
    int kk = k0 + half * 16 + j * 2;
    u.w[j] = *(const unsigned int*)(base + (long)m * sm + kk);
  }
  return u.v;
}

// K^T fragment: lane's 16 values are contiguous in head-dim -> 2x b128
__device__ inline v16bf load_kt_frag(const ushort_t* __restrict__ kt,
                                     int kd0, int n0) {
  int lane = threadIdx.x & 31;
  const ushort_t* p = kt + (long)(n0 + (lane & 15)) * kD + kd0 + (lane >> 4) * 16;
  FragQ u;
  u.q[0] = *(const uint4*)(p);
  u.q[1] = *(const uint4*)(p + 8);
  return u.v;
}

__device__ inline v16bf load_v_frag(const ushort_t* __restrict__ vt, int n0) {
  int lane = threadIdx.x & 31;
  int n = n0 + (lane & 15);
  int khalf = lane >> 4;
  FragQ u;
#pragma unroll
  for (int j = 0; j < 8; ++j) {
    int kk = khalf * 16 + 2 * j;
    u.s[2 * j]     = vt[(long)kk * kD + n];
    u.s[2 * j + 1] = vt[(long)(kk + 1) * kD + n];
  }
  return u.v;
}

// ---------------- fused flash attention (1 wave / 16 query rows) ----------------
// q,k,v,o bf16 [B, rows, D] with heads interleaved (D = H*DH)
__global__ __launch_bounds__(32) void attn_kernel(
    const ushort_t* __restrict__ q, const ushort_t* __restrict__ k,
    const ushort_t* __restrict__ v, ushort_t* __restrict__ o,
    int M, int Skv, float scale) {
  int b  = blockIdx.z;
  int h  = blockIdx.y;
  int m0 = blockIdx.x * 16;
  int lane  = threadIdx.x & 31;
  int mhalf = lane >> 4;
  int ncol  = lane & 15;

  const ushort_t* qbase = q + (long)b * M   * kD + h * kDH;
  const ushort_t* kbase = k + (long)b * Skv * kD + h * kDH;
  const ushort_t* vbase = v + (long)b * Skv * kD + h * kDH;

  v16bf aq0 = load_a_frag_h(qbase, m0, 0,  kD);
  v16bf aq1 = load_a_frag_h(qbase, m0, 32, kD);

  v8f oacc[4] = {v8f{}, v8f{}, v8f{}, v8f{}};
  float mcur[8], lsum[8];
#pragma unroll
  for (int r = 0; r < 8; ++r) { mcur[r] = -3.0e38f; lsum[r] = 0.f; }

  __shared__ __align__(8) ushort_t Pl[16 * 32];

  for (int t = 0; t < Skv; t += 32) {
    const ushort_t* kt = kbase + (long)t * kD;
    const ushort_t* vt = vbase + (long)t * kD;

    v8f s0 = {}, s1 = {};
    s0 = wmma_bf16(aq0, load_kt_frag(kt, 0, 0),   s0);
    s0 = wmma_bf16(aq1, load_kt_frag(kt, 32, 0),  s0);
    s1 = wmma_bf16(aq0, load_kt_frag(kt, 0, 16),  s1);
    s1 = wmma_bf16(aq1, load_kt_frag(kt, 32, 16), s1);

    __syncthreads();
#pragma unroll
    for (int r = 0; r < 8; ++r) {
      float a0 = s0[r] * scale, a1 = s1[r] * scale;
      float mx = fmaxf(a0, a1);
#pragma unroll
      for (int msk = 1; msk < 16; msk <<= 1) mx = fmaxf(mx, __shfl_xor(mx, msk, 32));
      float nm   = fmaxf(mcur[r], mx);
      float corr = __expf(mcur[r] - nm);
      float p0 = __expf(a0 - nm), p1 = __expf(a1 - nm);
      float ps = p0 + p1;
#pragma unroll
      for (int msk = 1; msk < 16; msk <<= 1) ps += __shfl_xor(ps, msk, 32);
      lsum[r] = lsum[r] * corr + ps;
      mcur[r] = nm;
#pragma unroll
      for (int c2 = 0; c2 < 4; ++c2) oacc[c2][r] *= corr;
      int prow = r + 8 * mhalf;
      Pl[prow * 32 + ncol]      = f32_bf16(p0);
      Pl[prow * 32 + 16 + ncol] = f32_bf16(p1);
    }
    __syncthreads();

    FragQ pu;
#pragma unroll
    for (int j = 0; j < 8; ++j) {
      int kk = mhalf * 16 + j * 2;
      pu.w[j] = *(const unsigned int*)(Pl + ncol * 32 + kk);
    }
#pragma unroll
    for (int c2 = 0; c2 < 4; ++c2)
      oacc[c2] = wmma_bf16(pu.v, load_v_frag(vt, c2 * 16), oacc[c2]);
  }

#pragma unroll
  for (int r = 0; r < 8; ++r) {
    float inv = 1.0f / lsum[r];
    int row = m0 + r + 8 * mhalf;
#pragma unroll
    for (int c2 = 0; c2 < 4; ++c2) {
      int col = h * kDH + c2 * 16 + ncol;
      o[((long)b * M + row) * kD + col] = f32_bf16(oacc[c2][r] * inv);
    }
  }
}

// ---------------- LayerNorm fp32 -> bf16 (1 wave / row of 512) ----------------
__global__ __launch_bounds__(256) void ln_kernel(const float* __restrict__ x,
                                                 ushort_t* __restrict__ y, int rows) {
  int wid = threadIdx.x >> 5, lane = threadIdx.x & 31;
  int row = blockIdx.x * 8 + wid;
  if (row >= rows) return;
  const float* xr = x + (long)row * kD;
  float vv[16];
  float s = 0.f, ss = 0.f;
#pragma unroll
  for (int i = 0; i < 16; ++i) {
    float t = xr[lane + i * 32];
    vv[i] = t; s += t; ss += t * t;
  }
#pragma unroll
  for (int m = 1; m < 32; m <<= 1) { s += __shfl_xor(s, m, 32); ss += __shfl_xor(ss, m, 32); }
  float mean = s * (1.f / kD);
  float var  = ss * (1.f / kD) - mean * mean;
  float inv  = rsqrtf(var + 1e-5f);
  ushort_t* yr = y + (long)row * kD;
#pragma unroll
  for (int i = 0; i < 16; ++i) yr[lane + i * 32] = f32_bf16((vv[i] - mean) * inv);
}

// ---------------- tokenize: fields -> tokens ----------------
__global__ void tokens_kernel(const float* __restrict__ fields,
                              const float* __restrict__ Wtok,
                              float* __restrict__ tokens) {
  long idx = (long)blockIdx.x * blockDim.x + threadIdx.x;
  if (idx >= (long)kB * kNT * kD) return;
  int d = (int)(idx & (kD - 1));
  int n = (int)((idx >> 9) & (kNT - 1));
  int b = (int)(idx >> 21);
  int h = n >> 6, w = n & 63;
  float acc = 0.f;
#pragma unroll
  for (int a = 0; a < kPH; ++a)
#pragma unroll
    for (int c = 0; c < kPW; ++c)
#pragma unroll
      for (int vv = 0; vv < kV; ++vv)
        acc += fields[(((long)b * 256 + (h * 4 + a)) * 256 + (w * 4 + c)) * kV + vv]
             * Wtok[(((a * kPW) + c) * kV + vv) * kD + d];
  tokens[idx] = acc;
}

// ---------------- gather + posemb -> context ----------------
__global__ __launch_bounds__(256) void build_context_kernel(
    const float* __restrict__ tokens, const int* __restrict__ srcs,
    const float* __restrict__ Wp, float* __restrict__ ctx) {
  int bs = blockIdx.x;
  int b = bs / kS, s = bs - b * kS;
  int n = srcs[(long)b * kS + s];
  float ch = (float)(n / kTW), cw = (float)(n % kTW);
  __shared__ float feats[64];
  int t = threadIdx.x;
  if (t < 64) {
    int grp = t >> 4, f = t & 15;
    float lam = exp2f(7.0f * (float)f / 15.0f);  // geomspace(1, 128, 16)
    float coo = (grp < 2) ? ch : cw;
    float ang = 6.283185307179586f * coo / lam;
    feats[t] = (grp & 1) ? cosf(ang) : sinf(ang);
  }
  __syncthreads();
  for (int d = t; d < kD; d += 256) {
    float acc = 0.f;
#pragma unroll 8
    for (int q = 0; q < 64; ++q) acc += feats[q] * Wp[q * kD + d];
    ctx[((long)b * kS + s) * kD + d] = tokens[((long)b * kNT + n) * kD + d] + acc;
  }
}

// ---------------- posemb -> queries ----------------
__global__ __launch_bounds__(256) void build_queries_kernel(
    const float* __restrict__ Wp, float* __restrict__ qout) {
  int bn = blockIdx.x;
  int b = bn / kNT, n = bn - b * kNT;
  float ch = (float)(n / kTW), cw = (float)(n % kTW);
  __shared__ float feats[64];
  int t = threadIdx.x;
  if (t < 64) {
    int grp = t >> 4, f = t & 15;
    float lam = exp2f(7.0f * (float)f / 15.0f);
    float coo = (grp < 2) ? ch : cw;
    float ang = 6.283185307179586f * coo / lam;
    feats[t] = (grp & 1) ? cosf(ang) : sinf(ang);
  }
  __syncthreads();
  for (int d = t; d < kD; d += 256) {
    float acc = 0.f;
#pragma unroll 8
    for (int q = 0; q < 64; ++q) acc += feats[q] * Wp[q * kD + d];
    qout[((long)b * kNT + n) * kD + d] = acc;
  }
}

// ---------------- broadcast latents ----------------
__global__ void init_z_kernel(const float* __restrict__ lat, float* __restrict__ z) {
  long idx = (long)blockIdx.x * blockDim.x + threadIdx.x;
  if (idx >= (long)kB * kNL * kD) return;
  z[idx] = lat[idx % ((long)kNL * kD)];
}

// ---------------- to_fields ----------------
__global__ void to_fields_kernel(const float* __restrict__ q,
                                 const float* __restrict__ Wfld,
                                 float* __restrict__ out) {
  long idx = (long)blockIdx.x * blockDim.x + threadIdx.x;
  const long total = (long)kB * 256 * 256 * kV * kK;
  if (idx >= total) return;
  long r = idx;
  int kk = (int)(r % kK); r /= kK;
  int vv = (int)(r % kV); r /= kV;
  int col = (int)(r % 256); r /= 256;
  int row = (int)(r % 256); int b = (int)(r / 256);
  int h = row >> 2, a = row & 3, w = col >> 2, c = col & 3;
  const float* tr = q + (((long)b * kNT) + h * kTW + w) * kD;
  const float* wp = Wfld + (((long)vv * kPH + a) * kPW + c) * kK + kk;
  float acc = 0.f;
  for (int d = 0; d < kD; ++d) acc += tr[d] * wp[(long)d * (kV * kPH * kPW * kK)];
  out[idx] = acc;
}

// ---------------- host orchestration ----------------
extern "C" void kernel_launch(void* const* d_in, const int* in_sizes, int n_in,
                              void* d_out, int out_size, void* d_ws, size_t ws_size,
                              hipStream_t stream) {
  (void)in_sizes; (void)n_in; (void)out_size; (void)ws_size;
  const float* fields   = (const float*)d_in[0];
  const float* Wtok     = (const float*)d_in[1];
  const float* Wfld     = (const float*)d_in[2];
  const float* Wpos_src = (const float*)d_in[3];
  const float* Wpos_tgt = (const float*)d_in[4];
  const float* latents0 = (const float*)d_in[5];
  const float* Wr_attn  = (const float*)d_in[6];
  const float* Wr_mlp1  = (const float*)d_in[7];
  const float* Wr_mlp2  = (const float*)d_in[8];
  const float* Wc_attn  = (const float*)d_in[9];
  const float* Wc_mlp1  = (const float*)d_in[10];
  const float* Wc_mlp2  = (const float*)d_in[11];
  const float* Ww_attn  = (const float*)d_in[12];
  const float* Ww_mlp1  = (const float*)d_in[13];
  const float* Ww_mlp2  = (const float*)d_in[14];
  const int*   srcs     = (const int*)d_in[15];
  float* out = (float*)d_out;

  char* wsb = (char*)d_ws;
  size_t off = 0;
  auto allocf = [&](size_t n) { float* p = (float*)(wsb + off);
                                off += ((n * 4 + 255) / 256) * 256; return p; };
  auto alloch = [&](size_t n) { ushort_t* p = (ushort_t*)(wsb + off);
                                off += ((n * 2 + 255) / 256) * 256; return p; };

  // fp32 activations
  float* tokens  = allocf((size_t)kB * kNT * kD);
  float* context = allocf((size_t)kB * kS  * kD);
  float* queries = allocf((size_t)kB * kNT * kD);
  float* z       = allocf((size_t)kB * kNL * kD);
  // bf16 activations
  ushort_t* lnA      = alloch((size_t)kB * kNT * kD);
  ushort_t* lnB      = alloch((size_t)kB * kS  * kD);
  ushort_t* qb       = alloch((size_t)kB * kNT * kD);
  ushort_t* kb       = alloch((size_t)kB * kS  * kD);
  ushort_t* vb       = alloch((size_t)kB * kS  * kD);
  ushort_t* attn_out = alloch((size_t)kB * kNT * kD);
  ushort_t* hb       = alloch((size_t)kB * kNT * kFF);
  // bf16 weights
  const long DD = (long)kD * kD;
  long nr_attn = (long)kL * 4 * DD, nc_attn = (long)kL * kCB * 4 * DD;
  long nr_mlp  = (long)kL * kD * kFF, nc_mlp = (long)kL * kCB * kD * kFF;
  ushort_t* Wr_attn_h = alloch(nr_attn);
  ushort_t* Wr_mlp1_h = alloch(nr_mlp);
  ushort_t* Wr_mlp2_h = alloch(nr_mlp);
  ushort_t* Wc_attn_h = alloch(nc_attn);
  ushort_t* Wc_mlp1_h = alloch(nc_mlp);
  ushort_t* Wc_mlp2_h = alloch(nc_mlp);
  ushort_t* Ww_attn_h = alloch(nr_attn);
  ushort_t* Ww_mlp1_h = alloch(nr_mlp);
  ushort_t* Ww_mlp2_h = alloch(nr_mlp);

  auto cvt = [&](const float* x, ushort_t* y, long n) {
    cvt_bf16_kernel<<<(unsigned)((n + 255) / 256), 256, 0, stream>>>(x, y, n);
  };
  cvt(Wr_attn, Wr_attn_h, nr_attn);
  cvt(Wr_mlp1, Wr_mlp1_h, nr_mlp);
  cvt(Wr_mlp2, Wr_mlp2_h, nr_mlp);
  cvt(Wc_attn, Wc_attn_h, nc_attn);
  cvt(Wc_mlp1, Wc_mlp1_h, nc_mlp);
  cvt(Wc_mlp2, Wc_mlp2_h, nc_mlp);
  cvt(Ww_attn, Ww_attn_h, nr_attn);
  cvt(Ww_mlp1, Ww_mlp1_h, nr_mlp);
  cvt(Ww_mlp2, Ww_mlp2_h, nr_mlp);

  auto gemm = [&](const ushort_t* A, const ushort_t* Bw, const float* R, void* C,
                  int M, int Nn, int K, long sA, long sB, long sR, long sC, int flags) {
    dim3 g(Nn / 64, M / 128, kB);
    gemm_wmma_kernel<<<g, 256, 0, stream>>>(A, Bw, R, C, M, Nn, K, sA, sB, sR, sC, flags);
  };
  auto ln = [&](const float* x, ushort_t* y, int rows) {
    ln_kernel<<<rows / 8, 256, 0, stream>>>(x, y, rows);
  };
  auto attn = [&](const ushort_t* q_, const ushort_t* k_, const ushort_t* v_,
                  ushort_t* o_, int M, int Skv) {
    dim3 g(M / 16, kH, kB);
    attn_kernel<<<g, 32, 0, stream>>>(q_, k_, v_, o_, M, Skv, 0.125f);
  };

  // ---- setup ----
  {
    long tot = (long)kB * kNT * kD;
    tokens_kernel<<<(unsigned)((tot + 255) / 256), 256, 0, stream>>>(fields, Wtok, tokens);
    build_context_kernel<<<kB * kS, 256, 0, stream>>>(tokens, srcs, Wpos_src, context);
    build_queries_kernel<<<kB * kNT, 256, 0, stream>>>(Wpos_tgt, queries);
    long zt = (long)kB * kNL * kD;
    init_z_kernel<<<(unsigned)((zt + 255) / 256), 256, 0, stream>>>(latents0, z);
  }

  const long zst = (long)kNL * kD, cst = (long)kS * kD, qst = (long)kNT * kD;

  for (int l = 0; l < kL; ++l) {
    // ---------------- read: latents x context ----------------
    {
      const ushort_t* Wq = Wr_attn_h + ((long)l * 4 + 0) * DD;
      const ushort_t* Wk = Wr_attn_h + ((long)l * 4 + 1) * DD;
      const ushort_t* Wv = Wr_attn_h + ((long)l * 4 + 2) * DD;
      const ushort_t* Wo = Wr_attn_h + ((long)l * 4 + 3) * DD;
      const ushort_t* W1 = Wr_mlp1_h + (long)l * kD * kFF;
      const ushort_t* W2 = Wr_mlp2_h + (long)l * kFF * kD;
      ln(z, lnA, kB * kNL);
      ln(context, lnB, kB * kS);
      gemm(lnA, Wq, nullptr, qb, kNL, kD, kD, zst, 0, 0, zst, 4);
      gemm(lnB, Wk, nullptr, kb, kS,  kD, kD, cst, 0, 0, cst, 4);
      gemm(lnB, Wv, nullptr, vb, kS,  kD, kD, cst, 0, 0, cst, 4);
      attn(qb, kb, vb, attn_out, kNL, kS);
      gemm(attn_out, Wo, z, z, kNL, kD, kD, zst, 0, zst, zst, 2);
      ln(z, lnA, kB * kNL);
      gemm(lnA, W1, nullptr, hb, kNL, kFF, kD, zst, 0, 0, (long)kNL * kFF, 5);
      gemm(hb, W2, z, z, kNL, kD, kFF, (long)kNL * kFF, 0, zst, zst, 2);
    }
    // ---------------- compute: latent self-attention ----------------
    for (int c = 0; c < kCB; ++c) {
      const ushort_t* Wq = Wc_attn_h + (((long)l * kCB + c) * 4 + 0) * DD;
      const ushort_t* Wk = Wc_attn_h + (((long)l * kCB + c) * 4 + 1) * DD;
      const ushort_t* Wv = Wc_attn_h + (((long)l * kCB + c) * 4 + 2) * DD;
      const ushort_t* Wo = Wc_attn_h + (((long)l * kCB + c) * 4 + 3) * DD;
      const ushort_t* W1 = Wc_mlp1_h + ((long)l * kCB + c) * kD * kFF;
      const ushort_t* W2 = Wc_mlp2_h + ((long)l * kCB + c) * kFF * kD;
      ln(z, lnA, kB * kNL);
      gemm(lnA, Wq, nullptr, qb, kNL, kD, kD, zst, 0, 0, zst, 4);
      gemm(lnA, Wk, nullptr, kb, kNL, kD, kD, zst, 0, 0, zst, 4);
      gemm(lnA, Wv, nullptr, vb, kNL, kD, kD, zst, 0, 0, zst, 4);
      attn(qb, kb, vb, attn_out, kNL, kNL);
      gemm(attn_out, Wo, z, z, kNL, kD, kD, zst, 0, zst, zst, 2);
      ln(z, lnA, kB * kNL);
      gemm(lnA, W1, nullptr, hb, kNL, kFF, kD, zst, 0, 0, (long)kNL * kFF, 5);
      gemm(hb, W2, z, z, kNL, kD, kFF, (long)kNL * kFF, 0, zst, zst, 2);
    }
    // ---------------- write: queries x latents ----------------
    {
      const ushort_t* Wq = Ww_attn_h + ((long)l * 4 + 0) * DD;
      const ushort_t* Wk = Ww_attn_h + ((long)l * 4 + 1) * DD;
      const ushort_t* Wv = Ww_attn_h + ((long)l * 4 + 2) * DD;
      const ushort_t* Wo = Ww_attn_h + ((long)l * 4 + 3) * DD;
      const ushort_t* W1 = Ww_mlp1_h + (long)l * kD * kFF;
      const ushort_t* W2 = Ww_mlp2_h + (long)l * kFF * kD;
      ln(queries, lnA, kB * kNT);
      ln(z, lnB, kB * kNL);
      gemm(lnA, Wq, nullptr, qb, kNT, kD, kD, qst, 0, 0, qst, 4);
      gemm(lnB, Wk, nullptr, kb, kNL, kD, kD, zst, 0, 0, zst, 4);
      gemm(lnB, Wv, nullptr, vb, kNL, kD, kD, zst, 0, 0, zst, 4);
      attn(qb, kb, vb, attn_out, kNT, kNL);
      gemm(attn_out, Wo, queries, queries, kNT, kD, kD, qst, 0, qst, qst, 2);
      ln(queries, lnA, kB * kNT);
      gemm(lnA, W1, nullptr, hb, kNT, kFF, kD, qst, 0, 0, (long)kNT * kFF, 5);
      gemm(hb, W2, queries, queries, kNT, kD, kFF, (long)kNT * kFF, 0, qst, qst, 2);
    }
  }

  // scatter with tgt=arange(N) replaces all tokens with queries -> read queries
  {
    long tot = (long)kB * 256 * 256 * kV * kK;
    to_fields_kernel<<<(unsigned)((tot + 255) / 256), 256, 0, stream>>>(queries, Wfld, out);
  }
}
